// Net_GAT_11227044511899
// MI455X (gfx1250) — matrix operand order repaired and checked by reading, not compiled, not run on hardware
//
#include <hip/hip_runtime.h>
#include <hip/hip_bf16.h>

// ---------------- problem constants (match reference) ----------------
#define NNODES 50000
#define NEDGES 800000
#define FIN    128
#define HEADS1 8
#define HIDDIM 256          // heads*out_ch for every layer
#define NGRAPH 512
#define NCLS   10

typedef __attribute__((ext_vector_type(8)))  __bf16 v8bf;
typedef __attribute__((ext_vector_type(16))) __bf16 v16bf;
typedef __attribute__((ext_vector_type(8)))  float  v8f;

static inline int cdiv(long a, long b) { return (int)((a + b - 1) / b); }

// ---------------- generic fill ----------------
__global__ void fill_kernel(float* __restrict__ p, float v, int n) {
  int i = blockIdx.x * blockDim.x + threadIdx.x;
  if (i < n) p[i] = v;
}

// ---------------- WMMA GEMM: C[M,Nc] = A[M,K] * B[K,Nc] ----------------
// fp32 in/out, bf16 multiply, fp32 accumulate.
// Block tile 64(M) x 64(N), 256 threads = 8 waves in 4(M) x 2(N);
// each wave computes a 16x32 strip (2 accumulators, shared A fragment).
// LDS layouts are fragment-major so every fragment is 2x ds_load_b128.
#define LDA 40   // As2 row stride in bf16 (80B, 16B aligned)
#define LDB 40   // Bs2 row stride in bf16

__global__ __launch_bounds__(256) void gemm_wmma_bf16(
    const float* __restrict__ A, const float* __restrict__ B,
    float* __restrict__ C, int M, int K, int Nc) {
  __shared__ __attribute__((aligned(16))) __bf16 As2[64][LDA]; // [m][k], k0..k0+31
  __shared__ __attribute__((aligned(16))) __bf16 Bs2[64][LDB]; // [n][k] (transposed)

  const int tid  = threadIdx.x;
  const int lane = tid & 31;
  const int wave = tid >> 5;
  const int wm   = wave & 3;           // 0..3  (M sub-tile, 16 rows each)
  const int wn   = wave >> 2;          // 0..1  (N strip, 32 cols each)
  const int bM   = blockIdx.y * 64;
  const int bN   = blockIdx.x * 64;

  v8f acc0 = {0.f,0.f,0.f,0.f,0.f,0.f,0.f,0.f};
  v8f acc1 = {0.f,0.f,0.f,0.f,0.f,0.f,0.f,0.f};

  for (int k0 = 0; k0 < K; k0 += 32) {
    // ---- stage A tile: 64 rows x 32 K, 8 floats/thread, one b128 LDS store ----
    {
      int r = tid >> 2;              // 0..63
      int c = (tid & 3) << 3;        // 0,8,16,24
      int gm = bM + r;
      float4 u = {0.f,0.f,0.f,0.f}, w = {0.f,0.f,0.f,0.f};
      if (gm < M) {
        const float* ap = A + (size_t)gm * K + k0 + c;
        u = *reinterpret_cast<const float4*>(ap);
        w = *reinterpret_cast<const float4*>(ap + 4);
      }
      v8bf av;
      av[0] = (__bf16)u.x; av[1] = (__bf16)u.y; av[2] = (__bf16)u.z; av[3] = (__bf16)u.w;
      av[4] = (__bf16)w.x; av[5] = (__bf16)w.y; av[6] = (__bf16)w.z; av[7] = (__bf16)w.w;
      *reinterpret_cast<v8bf*>(&As2[r][c]) = av;
    }
    // ---- stage B tile transposed: Bs2[n][k] = B[k0+k][bN+n] ----
    {
      int kk = tid >> 4;             // 0..15
      int n4 = (tid & 15) << 2;      // 0..60
      for (int kp = kk; kp < 32; kp += 16) {
        float4 u = *reinterpret_cast<const float4*>(B + (size_t)(k0 + kp) * Nc + bN + n4);
        Bs2[n4 + 0][kp] = (__bf16)u.x;
        Bs2[n4 + 1][kp] = (__bf16)u.y;
        Bs2[n4 + 2][kp] = (__bf16)u.z;
        Bs2[n4 + 3][kp] = (__bf16)u.w;
      }
    }
    __syncthreads();

    // ---- A fragment (16x32): lanes 0-15 K 0..15, lanes 16-31 K 16..31
    const int m  = wm * 16 + (lane & 15);
    const int ka = (lane < 16) ? 0 : 16;
    v8bf alo = *reinterpret_cast<const v8bf*>(&As2[m][ka]);
    v8bf ahi = *reinterpret_cast<const v8bf*>(&As2[m][ka + 8]);
    v16bf af = __builtin_shufflevector(alo, ahi, 0,1,2,3,4,5,6,7,8,9,10,11,12,13,14,15);

    // ---- B fragments (32x16 each): lanes 0-15 K 0..15, lanes 16-31 K 16..31
    const int kb = (lane < 16) ? 0 : 16;
    const int n0 = wn * 32 + (lane & 15);
    v8bf b0lo = *reinterpret_cast<const v8bf*>(&Bs2[n0][kb]);
    v8bf b0hi = *reinterpret_cast<const v8bf*>(&Bs2[n0][kb + 8]);
    v16bf bf0 = __builtin_shufflevector(b0lo, b0hi, 0,1,2,3,4,5,6,7,8,9,10,11,12,13,14,15);
    v8bf b1lo = *reinterpret_cast<const v8bf*>(&Bs2[n0 + 16][kb]);
    v8bf b1hi = *reinterpret_cast<const v8bf*>(&Bs2[n0 + 16][kb + 8]);
    v16bf bf1 = __builtin_shufflevector(b1lo, b1hi, 0,1,2,3,4,5,6,7,8,9,10,11,12,13,14,15);

    acc0 = __builtin_amdgcn_wmma_f32_16x16x32_bf16(false, af, false, bf0,
                                                   (short)0, acc0, false, false);
    acc1 = __builtin_amdgcn_wmma_f32_16x16x32_bf16(false, af, false, bf1,
                                                   (short)0, acc1, false, false);
    __syncthreads();
  }

  // ---- store: C/D layout lanes 0-15 -> M rows 0..7, lanes 16-31 -> rows 8..15
  const int gm0 = bM + wm * 16 + ((lane < 16) ? 0 : 8);
  const int gn0 = bN + wn * 32 + (lane & 15);
#pragma unroll
  for (int r = 0; r < 8; ++r) {
    int row = gm0 + r;
    if (row < M) {
      C[(size_t)row * Nc + gn0]      = acc0[r];
      C[(size_t)row * Nc + gn0 + 16] = acc1[r];
    }
  }
}

// ---------------- attention coefficients: out[n,h] = dot(H[n,h,:], att[h,:]) ----------------
__global__ __launch_bounds__(256) void attn_coef_kernel(
    const float* __restrict__ H, const float* __restrict__ att,
    float* __restrict__ out, int N, int heads, int C) {
  int unit = blockIdx.x * 8 + (threadIdx.x >> 5);
  int lane = threadIdx.x & 31;
  if (unit >= N * heads) return;
  int n = unit / heads, h = unit % heads;
  const float* hp = H + (size_t)n * heads * C + (size_t)h * C;
  const float* ap = att + (size_t)h * C;
  float s = 0.f;
  for (int c = lane; c < C; c += 32) s += hp[c] * ap[c];
#pragma unroll
  for (int off = 16; off >= 1; off >>= 1) s += __shfl_xor(s, off, 32);
  if (lane == 0) out[unit] = s;
}

// ---------------- float atomic max via int trick ----------------
__device__ inline void atomicMaxF(float* addr, float val) {
  if (val >= 0.f) atomicMax((int*)addr, __float_as_int(val));
  else            atomicMin((unsigned int*)addr, (unsigned int)__float_as_int(val));
}

// ---------------- edge pass 1: leaky-relu logits + segment max ----------------
__global__ void edge_logits_kernel(
    const int* __restrict__ src, const int* __restrict__ dst,
    const float* __restrict__ asrc, const float* __restrict__ adst,
    float* __restrict__ elog, float* __restrict__ emax,
    int E, int N, int heads) {
  int idx = blockIdx.x * blockDim.x + threadIdx.x;
  int total = (E + N) * heads;
  if (idx >= total) return;
  int h = idx % heads;
  int e = idx / heads;
  int s, d;
  if (e < E) { s = src[e]; d = dst[e]; } else { s = d = e - E; }
  float v = asrc[(size_t)s * heads + h] + adst[(size_t)d * heads + h];
  v = (v > 0.f) ? v : 0.2f * v;          // leaky_relu, slope 0.2
  elog[idx] = v;
  atomicMaxF(&emax[(size_t)d * heads + h], v);
}

// ---------------- edge pass 2: exp(e - max) and denom ----------------
__global__ void edge_exp_kernel(
    const int* __restrict__ dst, float* __restrict__ elog,
    const float* __restrict__ emax, float* __restrict__ denom,
    int E, int N, int heads) {
  int idx = blockIdx.x * blockDim.x + threadIdx.x;
  int total = (E + N) * heads;
  if (idx >= total) return;
  int h = idx % heads;
  int e = idx / heads;
  int d = (e < E) ? dst[e] : (e - E);
  float ex = __expf(elog[idx] - emax[(size_t)d * heads + h]);
  elog[idx] = ex;
  atomicAdd(&denom[(size_t)d * heads + h], ex);
}

// ---------------- edge pass 3: agg[dst] += alpha * H[src], 4 channels/thread ----------------
__global__ void edge_aggregate_kernel(
    const int* __restrict__ src, const int* __restrict__ dst,
    const float* __restrict__ H, const float* __restrict__ elog,
    const float* __restrict__ denom, float* __restrict__ agg,
    int E, int N, int heads, int Cper) {
  const int quads = HIDDIM / 4;                 // 64
  long idx = (long)blockIdx.x * blockDim.x + threadIdx.x;
  long total = (long)(E + N) * quads;
  if (idx >= total) return;
  int q = (int)(idx % quads);
  int e = (int)(idx / quads);
  int s, d;
  if (e < E) { s = src[e]; d = dst[e]; } else { s = d = e - E; }
  int c = q * 4;
  int h = c / Cper;
  float alpha = elog[(size_t)e * heads + h] / denom[(size_t)d * heads + h];
  float4 hv = *reinterpret_cast<const float4*>(H + (size_t)s * HIDDIM + c);
  float* ap = agg + (size_t)d * HIDDIM + c;
  atomicAdd(ap + 0, alpha * hv.x);
  atomicAdd(ap + 1, alpha * hv.y);
  atomicAdd(ap + 2, alpha * hv.z);
  atomicAdd(ap + 3, alpha * hv.w);
}

// ---------------- bias + ELU ----------------
__global__ void bias_elu_kernel(const float* __restrict__ agg,
                                const float* __restrict__ bias,
                                float* __restrict__ act, int N, int D) {
  int i = blockIdx.x * blockDim.x + threadIdx.x;
  if (i >= N * D) return;
  float v = agg[i] + bias[i % D];
  act[i] = (v > 0.f) ? v : (__expf(v) - 1.f);
}

// ---------------- global mean pool (sum + count) ----------------
__global__ void pool_sum_kernel(const float* __restrict__ act,
                                const int* __restrict__ batch,
                                float* __restrict__ g, float* __restrict__ cnt,
                                int N, int D) {
  int i = blockIdx.x * blockDim.x + threadIdx.x;
  if (i >= N * D) return;
  int n = i / D, c = i % D;
  int b = batch[n];
  atomicAdd(&g[(size_t)b * D + c], act[i]);
  if (c == 0) atomicAdd(&cnt[b], 1.0f);
}

// ---------------- MLP head + log_softmax, one block per graph ----------------
__global__ __launch_bounds__(256) void mlp_head_kernel(
    const float* __restrict__ g, const float* __restrict__ cnt,
    const float* __restrict__ lw1, const float* __restrict__ lb1,
    const float* __restrict__ lw2, const float* __restrict__ lb2,
    float* __restrict__ out) {
  __shared__ float gm[HIDDIM];
  __shared__ float hid[32];
  __shared__ float logits[NCLS];
  int b = blockIdx.x, t = threadIdx.x;
  float inv = 1.0f / fmaxf(cnt[b], 1.0f);
  gm[t] = g[(size_t)b * HIDDIM + t] * inv;
  __syncthreads();
  if (t < 32) {
    float s = lb1[t];
    for (int k = 0; k < HIDDIM; ++k) s += gm[k] * lw1[k * 32 + t];
    hid[t] = fmaxf(s, 0.f);
  }
  __syncthreads();
  if (t < NCLS) {
    float s = lb2[t];
    for (int k = 0; k < 32; ++k) s += hid[k] * lw2[k * NCLS + t];
    logits[t] = s;
  }
  __syncthreads();
  if (t == 0) {
    float mx = logits[0];
    for (int i = 1; i < NCLS; ++i) mx = fmaxf(mx, logits[i]);
    float se = 0.f;
    for (int i = 0; i < NCLS; ++i) se += __expf(logits[i] - mx);
    float lse = mx + __logf(se);
    for (int i = 0; i < NCLS; ++i) out[(size_t)b * NCLS + i] = logits[i] - lse;
  }
}

// ---------------- one GAT layer ----------------
static void run_gat_layer(const float* actIn, int Fin,
                          const float* W, const float* attS, const float* attD,
                          const float* bias, int heads,
                          const int* src, const int* dst,
                          float* H, float* ASRC, float* ADST, float* EMAX,
                          float* DEN, float* ELOG, float* AGG, float* actOut,
                          hipStream_t stream) {
  const int N = NNODES, E = NEDGES, D = HIDDIM;
  const int Cper = D / heads;

  // GEMM: H = actIn @ W   [N,Fin] x [Fin,256]
  dim3 ggrid(D / 64, cdiv(N, 64));
  gemm_wmma_bf16<<<ggrid, 256, 0, stream>>>(actIn, W, H, N, Fin, D);

  // attention coefficients
  int units = N * heads;
  attn_coef_kernel<<<cdiv(units, 8), 256, 0, stream>>>(H, attS, ASRC, N, heads, Cper);
  attn_coef_kernel<<<cdiv(units, 8), 256, 0, stream>>>(H, attD, ADST, N, heads, Cper);

  // init segment buffers
  fill_kernel<<<cdiv(units, 256), 256, 0, stream>>>(EMAX, -1e30f, units);
  fill_kernel<<<cdiv(units, 256), 256, 0, stream>>>(DEN, 0.f, units);
  fill_kernel<<<cdiv((long)N * D, 256), 256, 0, stream>>>(AGG, 0.f, N * D);

  int etot = (E + N) * heads;
  edge_logits_kernel<<<cdiv(etot, 256), 256, 0, stream>>>(src, dst, ASRC, ADST,
                                                          ELOG, EMAX, E, N, heads);
  edge_exp_kernel<<<cdiv(etot, 256), 256, 0, stream>>>(dst, ELOG, EMAX, DEN, E, N, heads);

  long atot = (long)(E + N) * (D / 4);
  edge_aggregate_kernel<<<cdiv(atot, 256), 256, 0, stream>>>(src, dst, H, ELOG, DEN,
                                                             AGG, E, N, heads, Cper);
  bias_elu_kernel<<<cdiv((long)N * D, 256), 256, 0, stream>>>(AGG, bias, actOut, N, D);
}

extern "C" void kernel_launch(void* const* d_in, const int* in_sizes, int n_in,
                              void* d_out, int out_size, void* d_ws, size_t ws_size,
                              hipStream_t stream) {
  const float* x    = (const float*)d_in[0];
  const int*   ei   = (const int*)d_in[1];
  const int*   batch= (const int*)d_in[2];
  const float* W1   = (const float*)d_in[3];
  const float* as1  = (const float*)d_in[4];
  const float* ad1  = (const float*)d_in[5];
  const float* b1   = (const float*)d_in[6];
  const float* W2   = (const float*)d_in[7];
  const float* as2  = (const float*)d_in[8];
  const float* ad2  = (const float*)d_in[9];
  const float* b2   = (const float*)d_in[10];
  const float* W3   = (const float*)d_in[11];
  const float* as3  = (const float*)d_in[12];
  const float* ad3  = (const float*)d_in[13];
  const float* b3   = (const float*)d_in[14];
  const float* lw1  = (const float*)d_in[15];
  const float* lb1  = (const float*)d_in[16];
  const float* lw2  = (const float*)d_in[17];
  const float* lb2  = (const float*)d_in[18];
  (void)in_sizes; (void)n_in; (void)out_size; (void)ws_size;

  const int* src = ei;
  const int* dst = ei + NEDGES;

  // workspace carve-up (floats)
  float* ws   = (float*)d_ws;
  float* H    = ws;                                        // N*256
  float* ACT  = H    + (size_t)NNODES * HIDDIM;            // N*256
  float* AGG  = ACT  + (size_t)NNODES * HIDDIM;            // N*256
  float* ASRC = AGG  + (size_t)NNODES * HIDDIM;            // N*8
  float* ADST = ASRC + (size_t)NNODES * HEADS1;            // N*8
  float* EMAX = ADST + (size_t)NNODES * HEADS1;            // N*8
  float* DEN  = EMAX + (size_t)NNODES * HEADS1;            // N*8
  float* ELOG = DEN  + (size_t)NNODES * HEADS1;            // (E+N)*8
  float* G    = ELOG + (size_t)(NEDGES + NNODES) * HEADS1; // 512*256
  float* CNT  = G    + (size_t)NGRAPH * HIDDIM;            // 512

  // layer 1: x[N,128] -> ACT[N,256], 8 heads x 32
  run_gat_layer(x, FIN, W1, as1, ad1, b1, HEADS1, src, dst,
                H, ASRC, ADST, EMAX, DEN, ELOG, AGG, ACT, stream);
  // layer 2: ACT -> ACT, 1 head x 256
  run_gat_layer(ACT, HIDDIM, W2, as2, ad2, b2, 1, src, dst,
                H, ASRC, ADST, EMAX, DEN, ELOG, AGG, ACT, stream);
  // layer 3
  run_gat_layer(ACT, HIDDIM, W3, as3, ad3, b3, 1, src, dst,
                H, ASRC, ADST, EMAX, DEN, ELOG, AGG, ACT, stream);

  // global mean pool + MLP head + log_softmax
  int gsz = NGRAPH * HIDDIM;
  fill_kernel<<<cdiv(gsz, 256), 256, 0, stream>>>(G, 0.f, gsz);
  fill_kernel<<<cdiv(NGRAPH, 256), 256, 0, stream>>>(CNT, 0.f, NGRAPH);
  pool_sum_kernel<<<cdiv((long)NNODES * HIDDIM, 256), 256, 0, stream>>>(
      ACT, batch, G, CNT, NNODES, HIDDIM);
  mlp_head_kernel<<<NGRAPH, 256, 0, stream>>>(G, CNT, lw1, lb1, lw2, lb2,
                                              (float*)d_out);
}
